// Conv2dImg2Col_48455821033956
// MI455X (gfx1250) — matrix operand order repaired and checked by reading, not compiled, not run on hardware
//
#include <hip/hip_runtime.h>
#include <stdint.h>

// ---------------------------------------------------------------------------
// Implicit-GEMM 3x3 conv for MI455X (gfx1250), fp32 via bf16x3 split WMMA.
//   GEMM: M = C_out(256), N = pixels(32*56*56=100352), K = 9*128 = 1152
//   K order = (r,s,ci): each 32-wide K chunk is one (r,s) + 32 consecutive ci.
// Block = 256 threads (8 wave32), tile = 128 co x 128 pixels, BK = 32.
// Double-buffered LDS holds packed bf16 hi/lo tiles (80KB dynamic LDS).
// Split uses v_perm_b32 truncation (hi) + v_cvt_pk_bf16_f32/v_perm (lo) to
// keep VALU low so it co-executes under the XDL WMMA pipeline.
// ---------------------------------------------------------------------------

typedef __attribute__((ext_vector_type(16))) __bf16 v16bf;
typedef __attribute__((ext_vector_type(8)))  float  v8f;

struct Q8 { uint4 a, b; };   // 32 bytes == sizeof(v16bf)

#define LDS_ROW   20                 // u32 per row: 16 data (32 bf16) + 4 pad (80B pitch)
#define REG_SZ    (128 * LDS_ROW)    // one 128-row region, in u32
#define OFF_A_HI  0
#define OFF_A_LO  (1 * REG_SZ)
#define OFF_B_HI  (2 * REG_SZ)
#define OFF_B_LO  (3 * REG_SZ)
#define BUF_SZ    (4 * REG_SZ)       // 10240 u32 = 40KB per buffer

// Pack {a1.hi16, a0.hi16} in one v_perm_b32.
__device__ __forceinline__ uint32_t pack_hi16(uint32_t u0, uint32_t u1) {
  return __builtin_amdgcn_perm(u1, u0, 0x07060302u);
}

// Truncation-based bf16 hi/lo split of two fp32 values, packed results.
__device__ __forceinline__ void split_pack(float a0, float a1,
                                           uint32_t& hi, uint32_t& lo) {
  const uint32_t u0 = __builtin_bit_cast(uint32_t, a0);
  const uint32_t u1 = __builtin_bit_cast(uint32_t, a1);
  hi = pack_hi16(u0, u1);                         // 1x v_perm_b32
  const float h0 = __builtin_bit_cast(float, u0 & 0xFFFF0000u);
  const float h1 = __builtin_bit_cast(float, u1 & 0xFFFF0000u);
  const float r0 = a0 - h0;                       // packs into v_pk_add_f32 neg
  const float r1 = a1 - h1;
#if __has_builtin(__builtin_amdgcn_cvt_pk_bf16_f32)
  auto pk = __builtin_amdgcn_cvt_pk_bf16_f32(r0, r1);
  lo = __builtin_bit_cast(uint32_t, pk);          // 1x v_cvt_pk_bf16_f32
#else
  lo = pack_hi16(__builtin_bit_cast(uint32_t, r0),
                 __builtin_bit_cast(uint32_t, r1));
#endif
}

// A-matrix (16x32 bf16) lane layout: lanes 0-15 -> K[0..7],K[16..23];
// lanes 16-31 -> K[8..15],K[24..31]. Rows stored as 32 bf16 contiguous.
__device__ __forceinline__ v16bf load_frag_a(const uint32_t* region, int row16, int lane) {
  const uint32_t* p = region + (row16 * 16 + (lane & 15)) * LDS_ROW;
  const int sel = (lane >> 4) << 2;              // 0 or 4 u32 (16B)
  Q8 q;
  q.a = *reinterpret_cast<const uint4*>(p + sel);
  q.b = *reinterpret_cast<const uint4*>(p + 8 + sel);
  return __builtin_bit_cast(v16bf, q);
}

// B-matrix (32x16 bf16) lane layout: lanes 0-15 -> K[0..7],K[16..23]; lanes 16-31 -> K[8..15],K[24..31].
__device__ __forceinline__ v16bf load_frag_b(const uint32_t* region, int row16, int lane) {
  const uint32_t* p = region + (row16 * 16 + (lane & 15)) * LDS_ROW + ((lane >> 4) << 2);
  Q8 q;
  q.a = *reinterpret_cast<const uint4*>(p);
  q.b = *reinterpret_cast<const uint4*>(p + 8);
  return __builtin_bit_cast(v16bf, q);
}

__device__ __forceinline__ v8f wmma_bf16(v16bf a, v16bf b, v8f c) {
  return __builtin_amdgcn_wmma_f32_16x16x32_bf16(false, a, false, b, (short)0, c,
                                                 false, false);
}

__global__ void __launch_bounds__(256)
conv3x3_igemm_bf16x3(const float* __restrict__ x, const float* __restrict__ w,
                     const float* __restrict__ bias, float* __restrict__ out)
{
  extern __shared__ __align__(16) uint32_t lds[];   // 2 * 40KB double buffer

  const int t    = threadIdx.x;
  const int lane = t & 31;
  const int wid  = t >> 5;
  const int row  = t & 127;          // loader row: co_local (A) / pix_local (B)
  const int jb   = (t >> 7) << 4;    // 0 or 16: which half of the K chunk

  const int pixBase = blockIdx.x * 128;   // 784 tiles, exact
  const int coBase  = blockIdx.y * 128;   // 2 tiles

  // Per-thread loader decode (fixed for whole kernel)
  const int pixL = pixBase + row;
  const int nL   = pixL / 3136;
  const int remL = pixL - nL * 3136;
  const int ohL  = remL / 56;
  const int owL  = remL - ohL * 56;
  const int coL  = coBase + row;

  const int waveM = wid & 1;    // co direction: 2 waves x 64 co (4 tiles)
  const int waveN = wid >> 1;   // pixel direction: 4 waves x 32 pix (2 tiles)

  float aReg[16], bReg[16];

  v8f acc[4][2];
  const v8f vzero = {0.f, 0.f, 0.f, 0.f, 0.f, 0.f, 0.f, 0.f};
  #pragma unroll
  for (int mt = 0; mt < 4; ++mt)
    #pragma unroll
    for (int nt = 0; nt < 2; ++nt) acc[mt][nt] = vzero;

  // K chunk kc in [0,36): rs = kc>>2 in [0,9), ci0 = (kc&3)*32
  auto load_chunk = [&](int kc) {
    const int rs  = kc >> 2;
    const int ci0 = ((kc & 3) << 5) + jb;
    const int r   = rs / 3;
    const int s   = rs - r * 3;
    // weights: w[co][ci][r][s], ci stride = 9 floats
    const float* wp = w + coL * 1152 + ci0 * 9 + rs;
    #pragma unroll
    for (int j = 0; j < 16; ++j) aReg[j] = wp[j * 9];
    // patches: x[n][ci][ih][iw], ci stride = 3136 floats; zero-pad OOB
    const int ih = ohL + r - 1;
    const int iw = owL + s - 1;
    if ((unsigned)ih < 56u && (unsigned)iw < 56u) {
      const float* xp = x + nL * 401408 + ci0 * 3136 + ih * 56 + iw;
      #pragma unroll
      for (int j = 0; j < 16; ++j) bReg[j] = xp[j * 3136];
    } else {
      #pragma unroll
      for (int j = 0; j < 16; ++j) bReg[j] = 0.0f;
    }
  };

  auto cvt_store = [&](int buf) {
    uint32_t* base = lds + buf * BUF_SZ;
    const int off  = row * LDS_ROW + (jb >> 1);     // +0 or +8 u32 in the row
    uint32_t ah[8], al[8], bh[8], bl[8];
    #pragma unroll
    for (int j = 0; j < 8; ++j) {
      split_pack(aReg[2*j], aReg[2*j+1], ah[j], al[j]);
      split_pack(bReg[2*j], bReg[2*j+1], bh[j], bl[j]);
    }
    *reinterpret_cast<uint4*>(base + OFF_A_HI + off)     = uint4{ah[0], ah[1], ah[2], ah[3]};
    *reinterpret_cast<uint4*>(base + OFF_A_HI + off + 4) = uint4{ah[4], ah[5], ah[6], ah[7]};
    *reinterpret_cast<uint4*>(base + OFF_A_LO + off)     = uint4{al[0], al[1], al[2], al[3]};
    *reinterpret_cast<uint4*>(base + OFF_A_LO + off + 4) = uint4{al[4], al[5], al[6], al[7]};
    *reinterpret_cast<uint4*>(base + OFF_B_HI + off)     = uint4{bh[0], bh[1], bh[2], bh[3]};
    *reinterpret_cast<uint4*>(base + OFF_B_HI + off + 4) = uint4{bh[4], bh[5], bh[6], bh[7]};
    *reinterpret_cast<uint4*>(base + OFF_B_LO + off)     = uint4{bl[0], bl[1], bl[2], bl[3]};
    *reinterpret_cast<uint4*>(base + OFF_B_LO + off + 4) = uint4{bl[4], bl[5], bl[6], bl[7]};
  };

  // Prologue: stage chunk 0
  load_chunk(0);
  cvt_store(0);
  __syncthreads();

  #pragma unroll 2
  for (int kc = 0; kc < 36; ++kc) {
    const int  cur  = kc & 1;
    const bool more = (kc + 1) < 36;
    if (more) load_chunk(kc + 1);           // global prefetch into VGPRs

    const uint32_t* buf = lds + cur * BUF_SZ;
    v16bf aHi[4], aLo[4], bHi[2], bLo[2];
    #pragma unroll
    for (int mt = 0; mt < 4; ++mt) {
      aHi[mt] = load_frag_a(buf + OFF_A_HI, waveM * 4 + mt, lane);
      aLo[mt] = load_frag_a(buf + OFF_A_LO, waveM * 4 + mt, lane);
    }
    #pragma unroll
    for (int nt = 0; nt < 2; ++nt) {
      bHi[nt] = load_frag_b(buf + OFF_B_HI, waveN * 2 + nt, lane);
      bLo[nt] = load_frag_b(buf + OFF_B_LO, waveN * 2 + nt, lane);
    }
    #pragma unroll
    for (int mt = 0; mt < 4; ++mt)
      #pragma unroll
      for (int nt = 0; nt < 2; ++nt) {
        v8f c = acc[mt][nt];
        c = wmma_bf16(aLo[mt], bHi[nt], c);   // low-order terms first
        c = wmma_bf16(aHi[mt], bLo[nt], c);
        c = wmma_bf16(aHi[mt], bHi[nt], c);
        acc[mt][nt] = c;
      }
    __syncthreads();                          // all reads of `cur` done
    if (more) {
      cvt_store((kc + 1) & 1);                // fill other buffer
      __syncthreads();                        // stores visible before next reads
    }
  }

  // Epilogue: C/D layout -> lanes = pixels (coalesced), VGPR v = co row (+8 upper half)
  #pragma unroll
  for (int nt = 0; nt < 2; ++nt) {
    const int pix = pixBase + waveN * 32 + nt * 16 + (lane & 15);
    const int n   = pix / 3136;
    const int rem = pix - n * 3136;
    const int pixPart = n * 802816 + rem;     // n*C_out*HW + (oh*W+ow)
    #pragma unroll
    for (int mt = 0; mt < 4; ++mt) {
      const int coRow = coBase + waveM * 64 + mt * 16 + ((lane >> 4) << 3);
      const v8f c = acc[mt][nt];
      #pragma unroll
      for (int v = 0; v < 8; ++v) {
        const int co = coRow + v;
        out[pixPart + co * 3136] = c[v] + bias[co];
      }
    }
  }
}

extern "C" void kernel_launch(void* const* d_in, const int* in_sizes, int n_in,
                              void* d_out, int out_size, void* d_ws, size_t ws_size,
                              hipStream_t stream) {
  (void)in_sizes; (void)n_in; (void)out_size; (void)d_ws; (void)ws_size;
  const float* x    = (const float*)d_in[0];
  const float* wgt  = (const float*)d_in[1];
  const float* bias = (const float*)d_in[2];
  float*       out  = (float*)d_out;

  dim3 grid(784, 2, 1);                          // 784 pixel tiles x 2 co tiles
  const size_t shmem = 2u * BUF_SZ * sizeof(uint32_t);   // 80 KB double buffer
  conv3x3_igemm_bf16x3<<<grid, 256, shmem, stream>>>(x, wgt, bias, out);
}
